// GNNBranch_36807869727435
// MI455X (gfx1250) — matrix operand order repaired and checked by reading, not compiled, not run on hardware
//
#include <hip/hip_runtime.h>
#include <hip/hip_bf16.h>

// ---------------------------------------------------------------------------
// GNN edge branch for MI455X (gfx1250, wave32, WMMA):
//   msg  = relu(concat(edge_attr, x[src]) @ W + b)   -> bf16 WMMA, f32 accum
//   out  = scatter_add(msg, dst)                     -> global_atomic_add_f32
// Hot kernel processes only FULL 64-edge chunks (no guards); a scalar tail
// kernel (launched only if nEdges % 64 != 0) covers the remainder.
// ---------------------------------------------------------------------------

typedef __attribute__((ext_vector_type(16))) __bf16   v16bf;
typedef __attribute__((ext_vector_type(8)))  float    v8f;
typedef __attribute__((ext_vector_type(4)))  float    f32x4;
typedef __attribute__((ext_vector_type(4)))  unsigned u32x4;

#define EF     64          // edge features
#define NF     128         // node features
#define OF     128         // out features
#define CHUNK  64          // edges per block iteration (4 m-tiles of 16)
#define KPAD   200         // bf16 elements per LDS row (400B stride -> conflict-free b128)
#define NBLK   2048        // persistent grid

// one v_cvt_pk_bf16_f32 instead of 4 integer ops per pair
static __device__ __forceinline__ unsigned pack2(float a, float b) {
  __hip_bfloat162 h = __float22bfloat162_rn(make_float2(a, b));
  unsigned u; __builtin_memcpy(&u, &h, 4);
  return u;
}

// Pack W[192,128] (row-major f32) into per-wave WMMA B-fragment order:
// dword id = ((t*6 + kk)*32 + lane)*8 + v  holds bf16 pair
//   { W[k][n], W[k+1][n] },  n = t*16 + lane%16,  k = kk*32 + (lane/16)*16 + 2v
__global__ void pack_w_kernel(const float* __restrict__ W, unsigned* __restrict__ Wp) {
  int id = blockIdx.x * 256 + threadIdx.x;
  if (id >= 8 * 6 * 32 * 8) return;
  int v    = id & 7;
  int lane = (id >> 3) & 31;
  int kk   = (id >> 8) % 6;
  int t    = id / (8 * 32 * 6);
  int n = t * 16 + (lane & 15);
  int k = kk * 32 + (lane >> 4) * 16 + 2 * v;
  Wp[id] = pack2(W[k * OF + n], W[(k + 1) * OF + n]);
}

__global__ void zero_kernel(f32x4* __restrict__ p, int n4) {
  int i = blockIdx.x * 256 + threadIdx.x;
  if (i < n4) p[i] = (f32x4){0.f, 0.f, 0.f, 0.f};
}

// Scalar fallback for the (at most 63) edges past the last full chunk.
__global__ void tail_kernel(const float* __restrict__ x, const int* __restrict__ ei,
                            const float* __restrict__ ea, const float* __restrict__ W,
                            const float* __restrict__ bias, float* __restrict__ out,
                            int nEdges, int eStart) {
  int nTail = nEdges - eStart;
  int idx = blockIdx.x * blockDim.x + threadIdx.x;
  if (idx >= nTail * OF) return;
  int e = eStart + idx / OF;
  int o = idx % OF;
  const float* arow = ea + (size_t)e * EF;
  const float* xrow = x + (size_t)(unsigned)ei[e] * NF;
  float acc = bias[o];
  for (int k = 0; k < EF; ++k) acc += arow[k] * W[k * OF + o];
  for (int k = 0; k < NF; ++k) acc += xrow[k] * W[(EF + k) * OF + o];
  acc = fmaxf(acc, 0.f);
  unsafeAtomicAdd(out + ((size_t)((unsigned)ei[nEdges + e] * OF + o)), acc);
}

__global__ __launch_bounds__(256) void gnn_kernel(
    const float* __restrict__ x,     // [nNodes, 128] f32 (L2-resident)
    const int*   __restrict__ ei,    // [2, nEdges] int32
    const float* __restrict__ ea,    // [nEdges, 64] f32 (streamed once, NT)
    const float* __restrict__ bias,  // [128] f32
    const unsigned* __restrict__ Wp, // packed bf16 B fragments
    float*       __restrict__ out,   // [nNodes, 128] f32 (pre-zeroed)
    int nEdges, int nFull)           // nFull = nEdges / CHUNK (all chunks complete)
{
  __shared__ __align__(16) unsigned lds[CHUNK * (KPAD / 2)];   // 25.6 KB

  const int tid   = threadIdx.x;
  const int wave  = tid >> 5;          // owns N-tile [wave*16, wave*16+16)
  const int lane  = tid & 31;
  const int khalf = lane >> 4;
  const int nloc  = lane & 15;

  union Frag { u32x4 u[2]; v16bf v; };

  // B panel (192x16 bf16) for this wave's N-tile: 6 k-steps x 8 dwords = 48 VGPRs
  Frag bfrag[6];
  const u32x4* wp4 = (const u32x4*)Wp;
#pragma unroll
  for (int kk = 0; kk < 6; ++kk) {
    const u32x4* bp = wp4 + ((size_t)(wave * 6 + kk) * 32 + lane) * 2;
    bfrag[kk].u[0] = bp[0];
    bfrag[kk].u[1] = bp[1];
  }
  const float bn = bias[wave * 16 + nloc];
  const unsigned colOff = (unsigned)(wave * 16 + nloc);  // lane's output column

  const int row = tid >> 2;            // 0..63: cooperative A staging
  const int seg = tid & 3;

  for (int chunk = blockIdx.x; chunk < nFull; chunk += gridDim.x) {
    const long ebase = (long)chunk * CHUNK;
    const long e_ld  = ebase + row;

    // ---- stage A = [edge_attr | x[src]] as bf16 into LDS (row-major, padded) ----
    {
      // edge_attr: 16 f32 per thread, non-temporal (pure stream)
      const f32x4* ap = (const f32x4*)(ea + e_ld * EF + seg * 16);
      f32x4 f0 = __builtin_nontemporal_load(ap + 0);
      f32x4 f1 = __builtin_nontemporal_load(ap + 1);
      f32x4 f2 = __builtin_nontemporal_load(ap + 2);
      f32x4 f3 = __builtin_nontemporal_load(ap + 3);
      u32x4* dq = (u32x4*)(lds + row * (KPAD / 2) + seg * 8);
      dq[0] = (u32x4){pack2(f0.x, f0.y), pack2(f0.z, f0.w),
                      pack2(f1.x, f1.y), pack2(f1.z, f1.w)};
      dq[1] = (u32x4){pack2(f2.x, f2.y), pack2(f2.z, f2.w),
                      pack2(f3.x, f3.y), pack2(f3.z, f3.w)};
      // x gather: 32 f32 per thread (hits L2/WGP$: x is only 25.6 MB + prefetch)
      const int s = ei[e_ld];
      const f32x4* xp = (const f32x4*)(x + (size_t)(unsigned)s * NF + seg * 32);
      u32x4* dx = (u32x4*)(lds + row * (KPAD / 2) + 32 + seg * 16);
#pragma unroll
      for (int j = 0; j < 4; ++j) {
        f32x4 g0 = xp[2 * j], g1 = xp[2 * j + 1];
        dx[j] = (u32x4){pack2(g0.x, g0.y), pack2(g0.z, g0.w),
                        pack2(g1.x, g1.y), pack2(g1.z, g1.w)};
      }
    }

    // Early-load next chunk's src index; its latency hides under the WMMAs,
    // and we use it after the compute phase to prefetch the x row.
    const long next_e = (long)(chunk + gridDim.x) * CHUNK + row;
    const bool has_next = next_e < (long)nFull * CHUNK;
    int s_next = 0;
    if (has_next) s_next = ei[next_e];

    __syncthreads();

    // ---- 4 m-tiles (16 edges each) x 6 k-steps of 32 -> 24 WMMAs / wave ----
#pragma unroll
    for (int mt = 0; mt < 4; ++mt) {
      // 8 consecutive dst indices for this half-wave, loaded as 2x b128
      // before the WMMAs so the load latency hides under them.
      const int* dp = ei + nEdges + ebase + mt * 16 + khalf * 8;
      int dsti[8];
      *(int4*)(&dsti[0]) = *(const int4*)(dp + 0);
      *(int4*)(&dsti[4]) = *(const int4*)(dp + 4);

      const unsigned* arow = lds + (mt * 16 + nloc) * (KPAD / 2);

      // software-pipelined A-fragment loads (one k-step ahead)
      Frag a0, a1;
      a0.u[0] = *(const u32x4*)(arow + khalf * 8);
      a0.u[1] = *(const u32x4*)(arow + khalf * 8 + 4);

      v8f acc = {0.f, 0.f, 0.f, 0.f, 0.f, 0.f, 0.f, 0.f};
#pragma unroll
      for (int kk = 0; kk < 6; ++kk) {
        Frag& cur = (kk & 1) ? a1 : a0;
        Frag& nxt = (kk & 1) ? a0 : a1;
        if (kk < 5) {
          nxt.u[0] = *(const u32x4*)(arow + (kk + 1) * 16 + khalf * 8);
          nxt.u[1] = *(const u32x4*)(arow + (kk + 1) * 16 + khalf * 8 + 4);
        }
        acc = __builtin_amdgcn_wmma_f32_16x16x32_bf16(
            false, cur.v, false, bfrag[kk].v, (short)0, acc, false, false);
      }

      // epilogue: bias + ReLU + scatter-add, branch-free.
      // 32-bit scatter offset (out is 25.6 MB) -> saddr-form atomics, no
      // per-element 64-bit address math.
#pragma unroll
      for (int r = 0; r < 8; ++r) {
        float v = fmaxf(acc[r] + bn, 0.f);
        unsigned off = (unsigned)dsti[r] * (unsigned)OF + colOff;
        unsafeAtomicAdd(out + (size_t)off, v);
      }
    }

    // warm near caches for the next chunk's staging phase
    if (has_next) {
      __builtin_prefetch(ea + next_e * EF + seg * 16, 0, 3);
      __builtin_prefetch(x + (size_t)(unsigned)s_next * NF + seg * 32, 0, 3);
    }

    __syncthreads();
  }
}

extern "C" void kernel_launch(void* const* d_in, const int* in_sizes, int n_in,
                              void* d_out, int out_size, void* d_ws, size_t ws_size,
                              hipStream_t stream) {
  const float* x    = (const float*)d_in[0];
  const int*   ei   = (const int*)d_in[1];
  const float* ea   = (const float*)d_in[2];
  const float* W    = (const float*)d_in[3];
  const float* bias = (const float*)d_in[4];
  float* out = (float*)d_out;

  unsigned* Wp = (unsigned*)d_ws;           // 12288 dwords = 48 KB of scratch

  const int nEdges = in_sizes[2] / EF;      // 800000
  const int nFull  = nEdges / CHUNK;        // 12500 complete chunks
  const int nTail  = nEdges - nFull * CHUNK;

  pack_w_kernel<<<48, 256, 0, stream>>>(W, Wp);

  int n4 = out_size / 4;
  zero_kernel<<<(n4 + 255) / 256, 256, 0, stream>>>((f32x4*)out, n4);

  if (nFull > 0) {
    int grid = nFull < NBLK ? nFull : NBLK;
    gnn_kernel<<<grid, 256, 0, stream>>>(x, ei, ea, bias, Wp, out, nEdges, nFull);
  }
  if (nTail > 0) {
    int nthr = nTail * OF;
    tail_kernel<<<(nthr + 127) / 128, 128, 0, stream>>>(x, ei, ea, W, bias, out,
                                                        nEdges, nFull * CHUNK);
  }
}
